// ActorCriticRNNCompat_42477226557883
// MI455X (gfx1250) — compile-verified, matrix operand order checked by reading, not run on hardware
//
#include <hip/hip_runtime.h>
#include <hip/hip_bf16.h>
#include <cmath>

// ---------------------------------------------------------------------------
// CDNA5 (gfx1250) wave32 WMMA helpers
// ---------------------------------------------------------------------------
typedef __attribute__((ext_vector_type(16))) _Float16 v16h;
typedef __attribute__((ext_vector_type(8)))  float    v8f;
typedef __attribute__((ext_vector_type(4)))  unsigned int u32x4;
typedef __attribute__((ext_vector_type(8)))  int      i32x8;
typedef __attribute__((ext_vector_type(4)))  int      i32x4;

// Tensor Data Mover availability (device pass only; host pass parses fallback)
#if defined(__has_builtin)
#  if __has_builtin(__builtin_amdgcn_tensor_load_to_lds) && \
      __has_builtin(__builtin_amdgcn_s_wait_tensorcnt)
#    define USE_TDM 1
#  endif
#endif
#ifndef USE_TDM
#  define USE_TDM 0
#endif

union FragU { v16h v; _Float16 h[16]; u32x4 q[2]; };

__device__ __forceinline__ v8f wmma_f16(v16h a, v16h b, v8f c) {
  // D = A(16x32 f16) * B(32x16 f16) + C(16x16 f32)
  return __builtin_amdgcn_wmma_f32_16x16x32_f16(false, a, false, b, (short)0, c,
                                                false, false);
}

// Load a 16x32 fragment from LDS. `base` points at row 0 of a row-major
// [16][32] f16 region (row stride 32 halves). Lane L<16 holds row L with
// K = {0..7, 16..23}; lane L+16 holds row L with K = {8..15, 24..31}.
__device__ __forceinline__ v16h ld_frag(const _Float16* base, int lane) {
  const int r  = lane & 15;
  const int lo = (lane >> 4) << 3;          // 0 or 8
  const char* p = (const char*)(base + r * 32);
  FragU u;
  u.q[0] = *(const u32x4*)(p + lo * 2);
  u.q[1] = *(const u32x4*)(p + (16 + lo) * 2);
  return u.v;
}

__device__ __forceinline__ float sigmoidf_(float x) {
  return 1.f / (1.f + __expf(-x));
}

#if USE_TDM
// Issue a TDM 2D tile load: global (tile1 rows x tile0 elems, row stride
// `stride` elems, 2-byte elements) -> LDS at byte address lds_addr, packed
// row-major [tile1][tile0]. Tracked by TENSORcnt.
__device__ __forceinline__ void tdm_load_2d(unsigned lds_addr, const void* g,
                                            unsigned tile0, unsigned tile1,
                                            unsigned stride, unsigned dim0,
                                            unsigned dim1) {
  const unsigned long long ga = (unsigned long long)g;
  // D# group 0: count=1 | lds_addr | global_addr[56:0] | type=2
  u32x4 g0 = { 1u,
               lds_addr,
               (unsigned)(ga & 0xffffffffu),
               (unsigned)((ga >> 32) & 0x1ffffffu) | (2u << 30) };
  // D# group 1: data_size=1 (2B); tensor_dim0/1; tile_dim0/1; dim0 stride
  i32x8 g1 = { (int)0x00010000u,
               (int)((dim0 & 0xffffu) << 16),
               (int)(((dim0 >> 16) & 0xffffu) | ((dim1 & 0xffffu) << 16)),
               (int)(((dim1 >> 16) & 0xffffu) | ((tile0 & 0xffffu) << 16)),
               (int)(tile1 & 0xffffu),
               (int)stride,
               0, 0 };
  i32x4 gz = { 0, 0, 0, 0 };
#if __clang_major__ >= 23
  i32x8 gz8 = { 0, 0, 0, 0, 0, 0, 0, 0 };
  __builtin_amdgcn_tensor_load_to_lds(g0, g1, gz, gz, gz8, 0);
#else
  __builtin_amdgcn_tensor_load_to_lds(g0, g1, gz, gz, 0);
#endif
}
#endif

// ---------------------------------------------------------------------------
// Conv weight prep: f32 HWIO -> f16 per-tap transposed image [kt][co][kk],
// channel-padded to 32 (kk >= Cin -> 0) so both the weights and the LDS
// feature maps use a uniform 32-channel K-tile layout.
// ---------------------------------------------------------------------------
__global__ void prepconvw(const float* __restrict__ k, _Float16* __restrict__ dst,
                          int ktiles, int Cin) {
  const long i = (long)blockIdx.x * 256 + threadIdx.x;
  if (i >= (long)ktiles * 1024) return;
  const int kt = (int)(i >> 10), r = (int)(i & 1023);
  const int co = r >> 5, kk = r & 31;
  dst[i] = (kk < Cin) ? (_Float16)k[((long)kt * Cin + kk) * 32 + co]
                      : (_Float16)0.f;
}

// ---------------------------------------------------------------------------
// Fused conv encoder: conv5x5(26->32) + conv3x3(32->32) + conv3x3(32->32),
// all SAME + ReLU, one 8x8 image per block (4 waves, M=64 pixels, N=32).
// Feature maps ping-pong between two LDS buffers; each stage's full weight
// set is staged in LDS once, so the tap loops are barrier-free.
// ---------------------------------------------------------------------------
__global__ __launch_bounds__(128) void convenc(
    const float* __restrict__ obs,
    const _Float16* __restrict__ wc1, const float* __restrict__ b1,
    const _Float16* __restrict__ wc2, const float* __restrict__ b2,
    const _Float16* __restrict__ wc3, const float* __restrict__ b3,
    _Float16* __restrict__ x3) {
  __shared__ alignas(16) _Float16 sW[25 * 1024];   // 50 KB: stage weights
  __shared__ alignas(16) _Float16 sB0[8 * 8 * 32]; // input (padded) / conv2 out
  __shared__ alignas(16) _Float16 sB1[8 * 8 * 32]; // conv1 out / conv2 in

  const long img  = blockIdx.x;
  const int  tid  = threadIdx.x, lane = tid & 31, wave = tid >> 5;
  const int  mBase = wave * 16;
  const int  mrow  = mBase + (lane & 15);
  const int  py = mrow >> 3, px = mrow & 7;
  const int  lo = (lane >> 4) << 3;

  const v8f vz = {};
  const u32x4 zq = {0u, 0u, 0u, 0u};

  // Stage padded input (ch 26..31 = 0) and all conv1 weights.
  const float* src = obs + img * (8 * 8 * 26);
  for (int i = tid; i < 2048; i += 128) {
    const int pix = i >> 5, c = i & 31;
    sB0[i] = (c < 26) ? (_Float16)src[pix * 26 + c] : (_Float16)0.f;
  }
  {
    const u32x4* g = (const u32x4*)wc1;
    u32x4* d = (u32x4*)sW;
    for (int i = tid; i < 25 * 128; i += 128) d[i] = g[i];
  }
  __syncthreads();

  // ---------------- conv1: 25 taps (5x5), 32-channel padded ----------------
  v8f a0 = vz, a1 = vz;
  for (int kt = 0; kt < 25; ++kt) {
    const int ky = kt / 5, kx = kt % 5;
    const int iy = py + ky - 2, ix = px + kx - 2;
    FragU a;
    if (iy >= 0 && iy < 8 && ix >= 0 && ix < 8) {
      const char* p = (const char*)(sB0 + (iy * 8 + ix) * 32);
      a.q[0] = *(const u32x4*)(p + lo * 2);
      a.q[1] = *(const u32x4*)(p + (16 + lo) * 2);
    } else { a.q[0] = zq; a.q[1] = zq; }
    v16h bf0 = ld_frag(&sW[kt * 1024],           lane);
    v16h bf1 = ld_frag(&sW[kt * 1024 + 16 * 32], lane);
    a0 = wmma_f16(a.v, bf0, a0);
    a1 = wmma_f16(a.v, bf1, a1);
  }
  {
    #pragma unroll
    for (int r = 0; r < 8; ++r) {
      const int m  = mBase + r + ((lane >> 4) << 3);
      const int c0 = lane & 15, c1 = 16 + (lane & 15);
      sB1[m * 32 + c0] = (_Float16)fmaxf(a0[r] + b1[c0], 0.f);
      sB1[m * 32 + c1] = (_Float16)fmaxf(a1[r] + b1[c1], 0.f);
    }
  }
  __syncthreads();   // conv1 done with sW/sB0; sB1 writes visible

  // ---------------- conv2: 9 taps (3x3) ------------------------------------
  {
    const u32x4* g = (const u32x4*)wc2;
    u32x4* d = (u32x4*)sW;
    for (int i = tid; i < 9 * 128; i += 128) d[i] = g[i];
  }
  __syncthreads();
  a0 = vz; a1 = vz;
  for (int kt = 0; kt < 9; ++kt) {
    const int ky = kt / 3, kx = kt % 3;
    const int iy = py + ky - 1, ix = px + kx - 1;
    FragU a;
    if (iy >= 0 && iy < 8 && ix >= 0 && ix < 8) {
      const char* p = (const char*)(sB1 + (iy * 8 + ix) * 32);
      a.q[0] = *(const u32x4*)(p + lo * 2);
      a.q[1] = *(const u32x4*)(p + (16 + lo) * 2);
    } else { a.q[0] = zq; a.q[1] = zq; }
    v16h bf0 = ld_frag(&sW[kt * 1024],           lane);
    v16h bf1 = ld_frag(&sW[kt * 1024 + 16 * 32], lane);
    a0 = wmma_f16(a.v, bf0, a0);
    a1 = wmma_f16(a.v, bf1, a1);
  }
  {
    #pragma unroll
    for (int r = 0; r < 8; ++r) {
      const int m  = mBase + r + ((lane >> 4) << 3);
      const int c0 = lane & 15, c1 = 16 + (lane & 15);
      sB0[m * 32 + c0] = (_Float16)fmaxf(a0[r] + b2[c0], 0.f);
      sB0[m * 32 + c1] = (_Float16)fmaxf(a1[r] + b2[c1], 0.f);
    }
  }
  __syncthreads();   // conv2 done with sW/sB1; sB0 (conv2 out) visible

  // ---------------- conv3: 9 taps (3x3) ------------------------------------
  {
    const u32x4* g = (const u32x4*)wc3;
    u32x4* d = (u32x4*)sW;
    for (int i = tid; i < 9 * 128; i += 128) d[i] = g[i];
  }
  __syncthreads();
  a0 = vz; a1 = vz;
  for (int kt = 0; kt < 9; ++kt) {
    const int ky = kt / 3, kx = kt % 3;
    const int iy = py + ky - 1, ix = px + kx - 1;
    FragU a;
    if (iy >= 0 && iy < 8 && ix >= 0 && ix < 8) {
      const char* p = (const char*)(sB0 + (iy * 8 + ix) * 32);
      a.q[0] = *(const u32x4*)(p + lo * 2);
      a.q[1] = *(const u32x4*)(p + (16 + lo) * 2);
    } else { a.q[0] = zq; a.q[1] = zq; }
    v16h bf0 = ld_frag(&sW[kt * 1024],           lane);
    v16h bf1 = ld_frag(&sW[kt * 1024 + 16 * 32], lane);
    a0 = wmma_f16(a.v, bf0, a0);
    a1 = wmma_f16(a.v, bf1, a1);
  }
  {
    #pragma unroll
    for (int r = 0; r < 8; ++r) {
      const int m  = mBase + r + ((lane >> 4) << 3);
      const int c0 = lane & 15, c1 = 16 + (lane & 15);
      x3[img * 2048 + m * 32 + c0] = (_Float16)fmaxf(a0[r] + b3[c0], 0.f);
      x3[img * 2048 + m * 32 + c1] = (_Float16)fmaxf(a1[r] + b3[c1], 0.f);
    }
  }
}

// ---------------------------------------------------------------------------
// Generic 128x128-tile f16 WMMA GEMM: C(f32) = act(A(f16) * B(f16) + bias).
// 256 threads = 8 waves, each wave computes 64x32 (4 M-tiles x 2 N-tiles).
// M, N multiples of 128; K multiple of 32. A tile staged via TDM (async
// tensor DMA) when available; B tile transposed cooperatively into LDS.
// ---------------------------------------------------------------------------
template <int RELU>
__global__ __launch_bounds__(256) void gemm128(
    const _Float16* __restrict__ A, int lda,
    const _Float16* __restrict__ B, int ldb,
    const float* __restrict__ bias,
    float* __restrict__ C, int ldc, int K) {
  __shared__ alignas(16) _Float16 As[128 * 32];
  __shared__ alignas(16) _Float16 Bs[128 * 32];   // transposed: [n][k]
  const int tid = threadIdx.x, lane = tid & 31;
  const int w = tid >> 5, wm = w & 1, wn = w >> 1;
  const long bM = (long)blockIdx.y * 128;
  const long bN = (long)blockIdx.x * 128;
  v8f acc[4][2] = {};

  for (int k0 = 0; k0 < K; k0 += 32) {
#if USE_TDM
    // Async DMA the 128x32 A tile straight into LDS (row-major, matches As).
    if (w == 0) {
      tdm_load_2d((unsigned)(size_t)&As[0], A + bM * lda + k0,
                  32u, 128u, (unsigned)lda, (unsigned)lda, 128u);
    }
#else
    {
      const int row = tid >> 1, off = (tid & 1) << 4;
      const u32x4* g = (const u32x4*)(A + (bM + row) * (long)lda + k0 + off);
      *(u32x4*)(As + row * 32 + off)     = g[0];
      *(u32x4*)(As + row * 32 + off + 8) = g[1];
    }
#endif
    {
      const int kk = tid >> 3, nb = tid & 7;
      const _Float16* g = B + (long)(k0 + kk) * ldb + bN + nb * 16;
      if (k0 + 32 < K)
        __builtin_prefetch(g + 32 * ldb, 0, 1);   // next K-tile -> L2/L0
      FragU t;
      t.q[0] = *(const u32x4*)g;
      t.q[1] = *(const u32x4*)(g + 8);
      #pragma unroll
      for (int i = 0; i < 16; ++i) Bs[(nb * 16 + i) * 32 + kk] = t.h[i];
    }
#if USE_TDM
    if (w == 0) __builtin_amdgcn_s_wait_tensorcnt(0);
#endif
    __syncthreads();
    #pragma unroll
    for (int nt = 0; nt < 2; ++nt) {
      v16h bf = ld_frag(&Bs[(wn * 32 + nt * 16) * 32], lane);
      #pragma unroll
      for (int mt = 0; mt < 4; ++mt) {
        v16h af = ld_frag(&As[(wm * 64 + mt * 16) * 32], lane);
        acc[mt][nt] = wmma_f16(af, bf, acc[mt][nt]);
      }
    }
    __syncthreads();
  }
  #pragma unroll
  for (int mt = 0; mt < 4; ++mt)
    #pragma unroll
    for (int nt = 0; nt < 2; ++nt)
      #pragma unroll
      for (int r = 0; r < 8; ++r) {
        const long row = bM + wm * 64 + mt * 16 + r + ((lane >> 4) << 3);
        const long col = bN + wn * 32 + nt * 16 + (lane & 15);
        float v = acc[mt][nt][r];
        if (bias) v += bias[col];
        if (RELU) v = fmaxf(v, 0.f);
        C[row * ldc + col] = v;
      }
}

// ---------------------------------------------------------------------------
// GRU step, part A: G = mask(h) @ [Wh_z | Wh_r]  (M=512, K=512, N=1024)
// Epilogue: z = sigmoid(Gz + wiz + bz) -> zbuf ; r = sigmoid(Gr + wir + br),
// rh = r * mask(h) (f16) for part B.
// ---------------------------------------------------------------------------
__global__ __launch_bounds__(256) void gru_a(
    const _Float16* __restrict__ h16, const _Float16* __restrict__ whzr,
    const unsigned char* __restrict__ done,
    const float* __restrict__ wi_t, const float* __restrict__ bz,
    const float* __restrict__ br, float* __restrict__ zbuf,
    _Float16* __restrict__ rh) {
  __shared__ alignas(16) _Float16 As[128 * 32];
  __shared__ alignas(16) _Float16 Bs[128 * 32];
  const int tid = threadIdx.x, lane = tid & 31;
  const int w = tid >> 5, wm = w & 1, wn = w >> 1;
  const int bM = blockIdx.y * 128, bN = blockIdx.x * 128;
  v8f acc[4][2] = {};

  for (int k0 = 0; k0 < 512; k0 += 32) {
    {
      const int row = tid >> 1, off = (tid & 1) << 4;
      u32x4 q0 = {0u, 0u, 0u, 0u}, q1 = q0;
      if (!done[bM + row]) {
        const u32x4* g = (const u32x4*)(h16 + (long)(bM + row) * 512 + k0 + off);
        q0 = g[0]; q1 = g[1];
      }
      *(u32x4*)(As + row * 32 + off)     = q0;
      *(u32x4*)(As + row * 32 + off + 8) = q1;
    }
    {
      const int kk = tid >> 3, nb = tid & 7;
      const _Float16* g = whzr + (long)(k0 + kk) * 1024 + bN + nb * 16;
      FragU t;
      t.q[0] = *(const u32x4*)g;
      t.q[1] = *(const u32x4*)(g + 8);
      #pragma unroll
      for (int i = 0; i < 16; ++i) Bs[(nb * 16 + i) * 32 + kk] = t.h[i];
    }
    __syncthreads();
    #pragma unroll
    for (int nt = 0; nt < 2; ++nt) {
      v16h bf = ld_frag(&Bs[(wn * 32 + nt * 16) * 32], lane);
      #pragma unroll
      for (int mt = 0; mt < 4; ++mt) {
        v16h af = ld_frag(&As[(wm * 64 + mt * 16) * 32], lane);
        acc[mt][nt] = wmma_f16(af, bf, acc[mt][nt]);
      }
    }
    __syncthreads();
  }
  #pragma unroll
  for (int mt = 0; mt < 4; ++mt)
    #pragma unroll
    for (int nt = 0; nt < 2; ++nt)
      #pragma unroll
      for (int r = 0; r < 8; ++r) {
        const int row = bM + wm * 64 + mt * 16 + r + ((lane >> 4) << 3);
        const int col = bN + wn * 32 + nt * 16 + (lane & 15);
        const float g = acc[mt][nt][r] + wi_t[(long)row * 1536 + col];
        if (col < 512) {
          zbuf[(long)row * 512 + col] = sigmoidf_(g + bz[col]);
        } else {
          const int c = col - 512;
          const float rr = sigmoidf_(g + br[c]);
          const float hm = done[row] ? 0.f : (float)h16[(long)row * 512 + c];
          rh[(long)row * 512 + c] = (_Float16)(rr * hm);
        }
      }
}

// ---------------------------------------------------------------------------
// GRU step, part B: Hh = (r*h) @ Wh_h  (M=512, K=512, N=512)
// Epilogue: h_hat = tanh(Hh + wih + bh); h = (1-z)*mask(h) + z*h_hat.
// Writes h (f32 + f16) and emb[t] (f16).
// ---------------------------------------------------------------------------
__global__ __launch_bounds__(256) void gru_b(
    const _Float16* __restrict__ rh, const _Float16* __restrict__ whh,
    const unsigned char* __restrict__ done, const float* __restrict__ wi_t,
    const float* __restrict__ bh, const float* __restrict__ zbuf,
    float* __restrict__ h32, _Float16* __restrict__ h16,
    _Float16* __restrict__ emb_t) {
  __shared__ alignas(16) _Float16 As[128 * 32];
  __shared__ alignas(16) _Float16 Bs[128 * 32];
  const int tid = threadIdx.x, lane = tid & 31;
  const int w = tid >> 5, wm = w & 1, wn = w >> 1;
  const int bM = blockIdx.y * 128, bN = blockIdx.x * 128;
  v8f acc[4][2] = {};

  for (int k0 = 0; k0 < 512; k0 += 32) {
#if USE_TDM
    if (w == 0) {
      tdm_load_2d((unsigned)(size_t)&As[0], rh + (long)bM * 512 + k0,
                  32u, 128u, 512u, 512u, 128u);
    }
#else
    {
      const int row = tid >> 1, off = (tid & 1) << 4;
      const u32x4* g = (const u32x4*)(rh + (long)(bM + row) * 512 + k0 + off);
      *(u32x4*)(As + row * 32 + off)     = g[0];
      *(u32x4*)(As + row * 32 + off + 8) = g[1];
    }
#endif
    {
      const int kk = tid >> 3, nb = tid & 7;
      const _Float16* g = whh + (long)(k0 + kk) * 512 + bN + nb * 16;
      FragU t;
      t.q[0] = *(const u32x4*)g;
      t.q[1] = *(const u32x4*)(g + 8);
      #pragma unroll
      for (int i = 0; i < 16; ++i) Bs[(nb * 16 + i) * 32 + kk] = t.h[i];
    }
#if USE_TDM
    if (w == 0) __builtin_amdgcn_s_wait_tensorcnt(0);
#endif
    __syncthreads();
    #pragma unroll
    for (int nt = 0; nt < 2; ++nt) {
      v16h bf = ld_frag(&Bs[(wn * 32 + nt * 16) * 32], lane);
      #pragma unroll
      for (int mt = 0; mt < 4; ++mt) {
        v16h af = ld_frag(&As[(wm * 64 + mt * 16) * 32], lane);
        acc[mt][nt] = wmma_f16(af, bf, acc[mt][nt]);
      }
    }
    __syncthreads();
  }
  #pragma unroll
  for (int mt = 0; mt < 4; ++mt)
    #pragma unroll
    for (int nt = 0; nt < 2; ++nt)
      #pragma unroll
      for (int r = 0; r < 8; ++r) {
        const int row = bM + wm * 64 + mt * 16 + r + ((lane >> 4) << 3);
        const int col = bN + wn * 32 + nt * 16 + (lane & 15);
        const float hh =
            tanhf(acc[mt][nt][r] + wi_t[(long)row * 1536 + 1024 + col] + bh[col]);
        const float z  = zbuf[(long)row * 512 + col];
        const float hm = done[row] ? 0.f : h32[(long)row * 512 + col];
        const float nh = (1.f - z) * hm + z * hh;
        h32[(long)row * 512 + col]   = nh;
        h16[(long)row * 512 + col]   = (_Float16)nh;
        emb_t[(long)row * 512 + col] = (_Float16)nh;
      }
}

// ---------------------------------------------------------------------------
// Row-wise LayerNorm over 512 features; one wave per row; f16 output.
// ---------------------------------------------------------------------------
__global__ __launch_bounds__(256) void lnk(const float* __restrict__ x,
                                           const float* __restrict__ sc,
                                           const float* __restrict__ bi,
                                           _Float16* __restrict__ y) {
  const long row = (long)blockIdx.x * 8 + (threadIdx.x >> 5);
  const int lane = threadIdx.x & 31;
  const float* xr = x + row * 512;
  float s = 0.f, s2 = 0.f;
  for (int i = lane; i < 512; i += 32) { const float v = xr[i]; s += v; s2 += v * v; }
  #pragma unroll
  for (int m = 16; m > 0; m >>= 1) {
    s  += __shfl_xor(s,  m, 32);
    s2 += __shfl_xor(s2, m, 32);
  }
  const float mu  = s  * (1.f / 512.f);
  const float var = s2 * (1.f / 512.f) - mu * mu;
  const float inv = rsqrtf(var + 1e-6f);
  for (int i = lane; i < 512; i += 32)
    y[row * 512 + i] = (_Float16)(((xr[i] - mu) * inv) * sc[i] + bi[i]);
}

// f32 -> f16 weight conversion with optional column offset (for concatenation)
__global__ void cvtk(const float* __restrict__ s, _Float16* __restrict__ d,
                     int cols, int ldd, int coff, long n) {
  const long i = (long)blockIdx.x * 256 + threadIdx.x;
  if (i >= n) return;
  const long r = i / cols, c = i % cols;
  d[r * ldd + coff + c] = (_Float16)s[i];
}

__global__ void inith(const float* __restrict__ h, float* __restrict__ h32,
                      _Float16* __restrict__ h16) {
  const int i = blockIdx.x * 256 + threadIdx.x;   // 512*512 elements
  const float v = h[i];
  h32[i] = v;
  h16[i] = (_Float16)v;
}

__global__ void copyh(const float* __restrict__ h32, float* __restrict__ out) {
  const int i = blockIdx.x * 256 + threadIdx.x;
  out[i] = h32[i];
}

// Tiny output heads: logits = relu_a @ (256x6), value = relu_c @ (256x1)
__global__ void headout(const float* __restrict__ aws, const float* __restrict__ cws,
                        const float* __restrict__ aok, const float* __restrict__ aob,
                        const float* __restrict__ cok, const float* __restrict__ cob,
                        float* __restrict__ logits, float* __restrict__ value) {
  const long row = (long)blockIdx.x * 256 + threadIdx.x;  // 32768 rows
  const float* ar = aws + row * 256;
  const float* cr = cws + row * 256;
  float l0 = aob[0], l1 = aob[1], l2 = aob[2], l3 = aob[3], l4 = aob[4], l5 = aob[5];
  float v = cob[0];
  for (int i = 0; i < 256; ++i) {
    const float a = ar[i];
    const float* wr = aok + i * 6;
    l0 += a * wr[0]; l1 += a * wr[1]; l2 += a * wr[2];
    l3 += a * wr[3]; l4 += a * wr[4]; l5 += a * wr[5];
    v += cr[i] * cok[i];
  }
  float* lp = logits + row * 6;
  lp[0] = l0; lp[1] = l1; lp[2] = l2; lp[3] = l3; lp[4] = l4; lp[5] = l5;
  value[row] = v;
}

// ---------------------------------------------------------------------------
// Host-side orchestration
// ---------------------------------------------------------------------------
extern "C" void kernel_launch(void* const* d_in, const int* in_sizes, int n_in,
                              void* d_out, int out_size, void* d_ws, size_t ws_size,
                              hipStream_t stream) {
  (void)in_sizes; (void)n_in; (void)out_size; (void)ws_size;

  const float* hidden = (const float*)d_in[0];
  const float* obs    = (const float*)d_in[1];
  const unsigned char* dones = (const unsigned char*)d_in[2];  // bool array
  const float* k1  = (const float*)d_in[3];  const float* b1  = (const float*)d_in[4];
  const float* k2  = (const float*)d_in[5];  const float* b2  = (const float*)d_in[6];
  const float* k3  = (const float*)d_in[7];  const float* b3  = (const float*)d_in[8];
  const float* dek = (const float*)d_in[9];  const float* deb = (const float*)d_in[10];
  const float* lns = (const float*)d_in[11]; const float* lnb = (const float*)d_in[12];
  const float* wiz = (const float*)d_in[13]; const float* wir = (const float*)d_in[14];
  const float* wih = (const float*)d_in[15];
  const float* whz = (const float*)d_in[16]; const float* whr = (const float*)d_in[17];
  const float* whh_ = (const float*)d_in[18];
  const float* bz  = (const float*)d_in[19]; const float* br  = (const float*)d_in[20];
  const float* bh  = (const float*)d_in[21];
  const float* afk = (const float*)d_in[22]; const float* afb = (const float*)d_in[23];
  const float* aok = (const float*)d_in[24]; const float* aob = (const float*)d_in[25];
  const float* cfk = (const float*)d_in[26]; const float* cfb = (const float*)d_in[27];
  const float* cok = (const float*)d_in[28]; const float* cob = (const float*)d_in[29];

  char* base = (char*)d_ws;
  size_t off = 0;
  auto alloc = [&](size_t bytes) -> void* {
    void* p = base + off;
    off = (off + bytes + 255) & ~(size_t)255;
    return p;
  };
  const size_t TN = 64ull * 512ull;  // 32768 rows

  _Float16* wdense = (_Float16*)alloc(2048ull * 512 * 2);
  _Float16* wicat  = (_Float16*)alloc(512ull * 1536 * 2);
  _Float16* whzr   = (_Float16*)alloc(512ull * 1024 * 2);
  _Float16* whhh   = (_Float16*)alloc(512ull * 512 * 2);
  _Float16* wafc   = (_Float16*)alloc(512ull * 256 * 2);
  _Float16* wcfc   = (_Float16*)alloc(512ull * 256 * 2);
  _Float16* wc1    = (_Float16*)alloc(25ull * 1024 * 2);
  _Float16* wc2    = (_Float16*)alloc(9ull * 1024 * 2);
  _Float16* wc3    = (_Float16*)alloc(9ull * 1024 * 2);
  _Float16* x3f    = (_Float16*)alloc(TN * 2048 * 2);
  float*    xws    = (float*)   alloc(TN * 512 * 4);
  _Float16* xn     = (_Float16*)alloc(TN * 512 * 2);
  float*    wi     = (float*)   alloc(TN * 1536 * 4);
  _Float16* emb    = (_Float16*)alloc(TN * 512 * 2);
  float*    aws    = (float*)   alloc(TN * 256 * 4);
  float*    cws    = (float*)   alloc(TN * 256 * 4);
  float*    h32    = (float*)   alloc(512ull * 512 * 4);
  _Float16* h16    = (_Float16*)alloc(512ull * 512 * 2);
  float*    zbuf   = (float*)   alloc(512ull * 512 * 4);
  _Float16* rh     = (_Float16*)alloc(512ull * 512 * 2);

  auto cv = [&](const float* s, _Float16* d, int rows, int cols, int ldd, int co) {
    const long n = (long)rows * cols;
    cvtk<<<dim3((unsigned)((n + 255) / 256)), dim3(256), 0, stream>>>(s, d, cols, ldd, co, n);
  };
  // Weight conversion / concatenation (f32 -> f16)
  cv(dek,  wdense, 2048, 512, 512,  0);
  cv(wiz,  wicat,  512,  512, 1536, 0);
  cv(wir,  wicat,  512,  512, 1536, 512);
  cv(wih,  wicat,  512,  512, 1536, 1024);
  cv(whz,  whzr,   512,  512, 1024, 0);
  cv(whr,  whzr,   512,  512, 1024, 512);
  cv(whh_, whhh,   512,  512, 512,  0);
  cv(afk,  wafc,   512,  256, 256,  0);
  cv(cfk,  wcfc,   512,  256, 256,  0);
  // Conv weights -> per-tap transposed, channel-padded f16 images
  prepconvw<<<dim3(100), dim3(256), 0, stream>>>(k1, wc1, 25, 26);
  prepconvw<<<dim3(36),  dim3(256), 0, stream>>>(k2, wc2, 9, 32);
  prepconvw<<<dim3(36),  dim3(256), 0, stream>>>(k3, wc3, 9, 32);

  inith<<<dim3(1024), dim3(256), 0, stream>>>(hidden, h32, h16);

  // Fused conv encoder: one image per block
  convenc<<<dim3(32768), dim3(128), 0, stream>>>(obs, wc1, b1, wc2, b2, wc3, b3, x3f);

  // Dense 2048 -> 512 + ReLU (M=32768)
  gemm128<1><<<dim3(4, 256), dim3(256), 0, stream>>>(x3f, 2048, wdense, 512, deb,
                                                     xws, 512, 2048);
  // LayerNorm -> f16
  lnk<<<dim3(4096), dim3(256), 0, stream>>>(xws, lns, lnb, xn);

  // Fused input projections: [wiz | wir | wih] (N=1536)
  gemm128<0><<<dim3(12, 256), dim3(256), 0, stream>>>(xn, 512, wicat, 1536, nullptr,
                                                      wi, 1536, 512);

  // Sequential GRU scan (64 steps x 2 dependent GEMMs)
  for (int t = 0; t < 64; ++t) {
    const float* wi_t = wi + (size_t)t * 512 * 1536;
    gru_a<<<dim3(8, 4), dim3(256), 0, stream>>>(h16, whzr, dones + (size_t)t * 512,
                                                wi_t, bz, br, zbuf, rh);
    gru_b<<<dim3(4, 4), dim3(256), 0, stream>>>(rh, whhh, dones + (size_t)t * 512,
                                                wi_t, bh, zbuf, h32, h16,
                                                emb + (size_t)t * 512 * 512);
  }

  // Heads (512 -> 256 + ReLU) over all T*N rows
  gemm128<1><<<dim3(2, 256), dim3(256), 0, stream>>>(emb, 512, wafc, 256, afb,
                                                     aws, 256, 512);
  gemm128<1><<<dim3(2, 256), dim3(256), 0, stream>>>(emb, 512, wcfc, 256, cfb,
                                                     cws, 256, 512);

  float* out = (float*)d_out;
  copyh<<<dim3(1024), dim3(256), 0, stream>>>(h32, out);                 // final hidden
  headout<<<dim3(128), dim3(256), 0, stream>>>(aws, cws, aok, aob, cok, cob,
                                               out + 512 * 512,          // logits
                                               out + 512 * 512 + TN * 6); // value
}
